// QLoRALinear_17927193493844
// MI455X (gfx1250) — matrix operand order repaired and checked by reading, not compiled, not run on hardware
//
#include <hip/hip_runtime.h>

typedef __attribute__((ext_vector_type(16))) _Float16 v16h;
typedef __attribute__((ext_vector_type(8)))  _Float16 v8h;
typedef __attribute__((ext_vector_type(2)))  _Float16 h2;
typedef __attribute__((ext_vector_type(8)))  float    v8f;
typedef __attribute__((ext_vector_type(4)))  float    f32x4;
typedef __attribute__((ext_vector_type(4)))  int      i32x4;
typedef __attribute__((ext_vector_type(8)))  int      i32x8;
typedef __attribute__((ext_vector_type(4)))  unsigned u32x4;

#define DIM_M   256      // B*S = 4*64
#define DIM_K   8192     // IN
#define DIM_N   8192     // OUT
#define RANK    16
#define BLK     64       // NF4 scale block == KT
#define MT      128
#define NT      64
#define KT      64
#define NKT     (DIM_K / KT)     // 128 k-tiles
#define SCALING 2.0f     // alpha/rank = 32/16

// NF4 levels: Phi^-1((i+0.5)/16) / Phi^-1(15.5/16)
__device__ __constant__ float kNF4[16] = {
    -1.00000f, -0.70757f, -0.54222f, -0.41682f,
    -0.31090f, -0.21597f, -0.12734f, -0.04209f,
     0.04209f,  0.12734f,  0.21597f,  0.31090f,
     0.41682f,  0.54222f,  0.70757f,  1.00000f
};

static __device__ __forceinline__ unsigned pack2h(float a, float b) {
    union { _Float16 h[2]; unsigned u; } p;
    p.h[0] = (_Float16)a;   // lower 16 bits -> even k (hi nibble)
    p.h[1] = (_Float16)b;   // upper 16 bits -> odd k (lo nibble)
    return p.u;
}

// ---------------------------------------------------------------------------
// TDM: 2D f16 tile load, MT rows x KT cols, row stride DIM_K, into LDS.
// D# layout per CDNA5 ISA.
// ---------------------------------------------------------------------------
static __device__ __forceinline__ void tdm_load_xtile(const _Float16* gptr,
                                                      unsigned lds_off) {
    unsigned long long ga = (unsigned long long)(uintptr_t)gptr;
    u32x4 g0;
    g0[0] = 1u;                                        // count=1 (valid D#)
    g0[1] = lds_off;                                   // lds_addr (bytes)
    g0[2] = (unsigned)(ga & 0xFFFFFFFFull);            // global_addr[31:0]
    g0[3] = (unsigned)((ga >> 32) & 0x01FFFFFFull)     // global_addr[56:32]
          | 0x80000000u;                               // type=2 ("image")
    i32x8 g1;
    g1[0] = 0x00010000;                 // data_size=1 (2B), wg_mask=0, no pad
    g1[1] = (int)((DIM_K & 0xFFFF) << 16);             // tensor_dim0[15:0]
    g1[2] = (int)(((DIM_K >> 16) & 0xFFFF)             // tensor_dim0[31:16]
          | ((DIM_M & 0xFFFF) << 16));                 // tensor_dim1[15:0]
    g1[3] = (int)(((DIM_M >> 16) & 0xFFFF)             // tensor_dim1[31:16]
          | (KT << 16));                               // tile_dim0 = 64
    g1[4] = MT;                                        // tile_dim1 = 128
    g1[5] = DIM_K;                                     // tensor_dim0_stride lo32
    g1[6] = 0;                                         // stride hi16 | dim1_stride lo
    g1[7] = 0;
    i32x4 z4 = {0, 0, 0, 0};
#if defined(__clang_major__) && (__clang_major__ >= 23)
    i32x8 z8 = {0, 0, 0, 0, 0, 0, 0, 0};
    __builtin_amdgcn_tensor_load_to_lds(g0, g1, z4, z4, z8, 0);
#else
    __builtin_amdgcn_tensor_load_to_lds(g0, g1, z4, z4, 0);
#endif
}

// ---------------------------------------------------------------------------
// Kernel 0: xh[m][k] = (f16) x[m][k]   (one-time fp32 -> f16 copy)
// ---------------------------------------------------------------------------
__global__ __launch_bounds__(256) void xcast_kernel(
    const float* __restrict__ x, _Float16* __restrict__ xh)
{
    const size_t i = ((size_t)blockIdx.x * 256 + threadIdx.x) * 8;
    f32x4 a = *(const f32x4*)(x + i);
    f32x4 b = *(const f32x4*)(x + i + 4);
    i32x4 o;
    o[0] = (int)pack2h(a[0], a[1]);
    o[1] = (int)pack2h(a[2], a[3]);
    o[2] = (int)pack2h(b[0], b[1]);
    o[3] = (int)pack2h(b[2], b[3]);
    *(i32x4*)(xh + i) = o;
}

// ---------------------------------------------------------------------------
// Kernel 1: t[m][r] = SCALING * sum_k x[m][k] * lora_A[r][k]   (256 x 16)
// ---------------------------------------------------------------------------
__global__ __launch_bounds__(256) void lora_t_kernel(
    const float* __restrict__ x, const float* __restrict__ lora_A,
    float* __restrict__ t)
{
    __shared__ float red[256 * RANK];   // 16 KB
    const int m   = blockIdx.x;
    const int tid = threadIdx.x;
    const float* xr = x + (size_t)m * DIM_K;

    float acc[RANK];
#pragma unroll
    for (int r = 0; r < RANK; ++r) acc[r] = 0.f;

    for (int k = tid; k < DIM_K; k += 256) {
        float xv = xr[k];
#pragma unroll
        for (int r = 0; r < RANK; ++r)
            acc[r] += xv * lora_A[(size_t)r * DIM_K + k];
    }
#pragma unroll
    for (int r = 0; r < RANK; ++r) red[tid * RANK + r] = acc[r];
    __syncthreads();

    for (int off = 128; off > 0; off >>= 1) {
        if (tid < off) {
#pragma unroll
            for (int r = 0; r < RANK; ++r)
                red[tid * RANK + r] += red[(tid + off) * RANK + r];
        }
        __syncthreads();
    }
    if (tid < RANK) t[m * RANK + tid] = SCALING * red[tid];
}

// ---------------------------------------------------------------------------
// Kernel 2: fused NF4-dequant GEMM + LoRA tail, WMMA f32_16x16x32_f16.
// Double-buffered LDS; x tiles via Tensor Data Mover (TENSORcnt), weights
// dequantized with a 256-entry packed-half2 table + v_pk_mul_f16.
// grid (DIM_N/NT, DIM_M/MT), 256 threads (8 wave32).
// ---------------------------------------------------------------------------
__global__ __launch_bounds__(256) void qlora_gemm_kernel(
    const _Float16* __restrict__ xh, const int* __restrict__ qw,
    const float* __restrict__ scales, const float* __restrict__ loraB,
    const float* __restrict__ t, float* __restrict__ out)
{
    __shared__ __align__(16) _Float16 xs[2][MT * KT];  // 2 x 16 KB, [m][k]
    __shared__ __align__(16) _Float16 ws[2][NT * KT];  // 2 x  8 KB, [n][k]
    __shared__ unsigned tab[256];                      // 1 KB packed NF4 pairs

    const int tid = threadIdx.x;
    const int w   = tid >> 5;          // wave 0..7 -> m-strip
    const int l   = tid & 31;
    const int h   = l >> 4;            // lane half
    const int ln  = l & 15;
    const int n0  = blockIdx.x * NT;
    const int m0  = blockIdx.y * MT;

    // byte -> {nf4[hi], nf4[lo]} as packed f16x2
    tab[tid] = pack2h(kNF4[(tid >> 4) & 15], kNF4[tid & 15]);

    v8f acc[4];
#pragma unroll
    for (int j = 0; j < 4; ++j) acc[j] = (v8f)(0.f);

    // weight-tile mapping: 4 threads per n-row, 8 packed int32 (16 k) each
    const int wrow = tid >> 2, wseg = tid & 3;
    const int* qbase = qw + (size_t)(n0 + wrow) * (DIM_K / 2) + wseg * 8;
    const float* scbase = scales + (size_t)(n0 + wrow) * (DIM_K / BLK);
    const unsigned lds_xs0 = (unsigned)(uintptr_t)&xs[0][0];
    const _Float16* xgbase = xh + (size_t)m0 * DIM_K;

    union AFrag { v16h v; v8h h8[2]; };
    union HU { unsigned u; h2 h; };

    // ---- prologue: prefetch tile 0 ----
    i32x4 qa = *(const i32x4*)(qbase);
    i32x4 qb = *(const i32x4*)(qbase + 4);
    float  sc = scbase[0];
    if (tid < 32) tdm_load_xtile(xgbase, lds_xs0);
    __syncthreads();                    // tab[] ready before first commit

    for (int i = 0; i < NKT; ++i) {
        const int buf = i & 1;
        // ---- commit weight tile i: table lookup + pk_mul -> LDS ----
        {
            HU sv; sv.h = (h2){(_Float16)sc, (_Float16)sc};
            i32x4 o0, o1;
#pragma unroll
            for (int c = 0; c < 4; ++c) {
                HU e; e.u = tab[qa[c] & 255]; e.h *= sv.h; o0[c] = (int)e.u;
            }
#pragma unroll
            for (int c = 0; c < 4; ++c) {
                HU e; e.u = tab[qb[c] & 255]; e.h *= sv.h; o1[c] = (int)e.u;
            }
            i32x4* wd = (i32x4*)&ws[buf][wrow * KT] + wseg * 2;
            wd[0] = o0;
            wd[1] = o1;
        }
        // ---- wait tile i's TDM, then launch tile i+1 prefetch ----
        if (tid < 32) __builtin_amdgcn_s_wait_tensorcnt(0);
        if (i + 1 < NKT) {
            const int kt = (i + 1) * KT;
            qa = *(const i32x4*)(qbase + (kt >> 1));
            qb = *(const i32x4*)(qbase + (kt >> 1) + 4);
            sc = scbase[kt / BLK];
            if (tid < 32)
                tdm_load_xtile(xgbase + kt, lds_xs0 + (unsigned)(buf ^ 1) * (MT * KT * 2));
        }
        __syncthreads();

        // ---- compute tile i: 8x WMMA ----
#pragma unroll
        for (int kk = 0; kk < KT; kk += 32) {
            AFrag a;
            const int abase = (w * 16 + ln) * KT + kk + 16 * h;
            a.h8[0] = *(const v8h*)&xs[buf][abase];
            a.h8[1] = *(const v8h*)&xs[buf][abase + 8];
#pragma unroll
            for (int j = 0; j < 4; ++j) {
                const int bbase = (j * 16 + ln) * KT + kk + 16 * h;
                v16h b = *(const v16h*)&ws[buf][bbase];
                acc[j] = __builtin_amdgcn_wmma_f32_16x16x32_f16(
                    false, a.v, false, b, (short)0, acc[j], false, false);
            }
        }
        __syncthreads();
    }

    // ---- LoRA tail: out += [2*x*A^T] @ [lora_B]^T as one extra K=32 tile ----
    if (tid < MT) {                       // xs rows: t (cols 0..15), zeros 16..31
        const float* tp = t + (size_t)(m0 + tid) * RANK;
        unsigned* xd = (unsigned*)&xs[0][tid * KT];
#pragma unroll
        for (int c = 0; c < 8; ++c) xd[c] = pack2h(tp[2 * c], tp[2 * c + 1]);
#pragma unroll
        for (int c = 8; c < 16; ++c) xd[c] = 0u;
    } else if (tid < MT + NT) {           // ws rows: lora_B (cols 0..15), zeros
        const int row = tid - MT;
        const float* bp = loraB + (size_t)(n0 + row) * RANK;
        unsigned* wd = (unsigned*)&ws[0][row * KT];
#pragma unroll
        for (int c = 0; c < 8; ++c) wd[c] = pack2h(bp[2 * c], bp[2 * c + 1]);
#pragma unroll
        for (int c = 8; c < 16; ++c) wd[c] = 0u;
    }
    __syncthreads();
    {
        AFrag a;
        const int abase = (w * 16 + ln) * KT + 16 * h;
        a.h8[0] = *(const v8h*)&xs[0][abase];
        a.h8[1] = *(const v8h*)&xs[0][abase + 8];
#pragma unroll
        for (int j = 0; j < 4; ++j) {
            const int bbase = (j * 16 + ln) * KT + 16 * h;
            v16h b = *(const v16h*)&ws[0][bbase];
            acc[j] = __builtin_amdgcn_wmma_f32_16x16x32_f16(
                false, a.v, false, b, (short)0, acc[j], false, false);
        }
    }

    // ---- store: C/D layout lane=(N), vgpr r=(M) ----
#pragma unroll
    for (int j = 0; j < 4; ++j) {
        const int gn = n0 + j * 16 + ln;
#pragma unroll
        for (int r = 0; r < 8; ++r) {
            const int gm = m0 + w * 16 + 8 * h + r;
            out[(size_t)gm * DIM_N + gn] = acc[j][r];
        }
    }
}

extern "C" void kernel_launch(void* const* d_in, const int* in_sizes, int n_in,
                              void* d_out, int out_size, void* d_ws, size_t ws_size,
                              hipStream_t stream) {
    const float* x      = (const float*)d_in[0];
    const int*   qw     = (const int*)  d_in[1];
    const float* scales = (const float*)d_in[2];
    const float* lora_A = (const float*)d_in[3];
    const float* lora_B = (const float*)d_in[4];
    float*       out    = (float*)d_out;

    float*     t  = (float*)d_ws;                       // 16 KB
    _Float16*  xh = (_Float16*)((char*)d_ws + 16384);   // 4 MiB f16 copy of x

    xcast_kernel<<<(DIM_M * DIM_K) / (256 * 8), 256, 0, stream>>>(x, xh);
    lora_t_kernel<<<DIM_M, 256, 0, stream>>>(x, lora_A, t);

    dim3 grid(DIM_N / NT, DIM_M / MT);
    qlora_gemm_kernel<<<grid, 256, 0, stream>>>(xh, qw, scales, lora_B, t, out);
}
